// EncoderBlock_76957224010036
// MI455X (gfx1250) — matrix operand order repaired and checked by reading, not compiled, not run on hardware
//
#include <hip/hip_runtime.h>
#include <hip/hip_bf16.h>

// ---------------- problem constants ----------------
constexpr int Bc  = 2;
constexpr int Lc  = 2048;
constexpr int Dc  = 1024;
constexpr int Hc  = 16;
constexpr int HDc = 64;       // Dc / Hc
constexpr int DFFc = 4096;
constexpr int NDELTA = 2 * Lc - 1;   // 4095 relative positions

typedef unsigned short u16;
typedef unsigned int   u32;

typedef __attribute__((ext_vector_type(16))) __bf16 v16bf;
typedef __attribute__((ext_vector_type(8)))  float  v8f;

struct __align__(16) U16B { u32 x, y, z, w; };   // 16-byte move unit

union Frag {                 // one WMMA A/B operand: 16 bf16 = 8 VGPRs
    v16bf v;
    U16B  q[2];
};

// round-to-nearest-even f32 -> bf16 bits
__device__ __forceinline__ u16 f2bf(float f) {
    u32 u = __float_as_uint(f);
    u += 0x7FFFu + ((u >> 16) & 1u);
    return (u16)(u >> 16);
}

// gfx1250 async copy: global -> LDS, 16B per lane, tracked by ASYNCcnt.
// Bypasses VGPRs entirely (the CDNA5 path for producer->LDS staging).
__device__ __forceinline__ void async_copy_b128(void* lds_ptr, const void* gptr) {
    u32 lds_off = (u32)(size_t)lds_ptr;          // addrspace(3) offset
    unsigned long long ga = (unsigned long long)gptr;
    asm volatile("global_load_async_to_lds_b128 %0, %1, off"
                 :: "v"(lds_off), "v"(ga) : "memory");
}
__device__ __forceinline__ void wait_async0() {
    asm volatile("s_wait_asynccnt 0x0" ::: "memory");
}

// ---------------- weight / activation cast ----------------
__global__ void cast_bf16_kernel(const float* __restrict__ in, u16* __restrict__ out, int n) {
    int i = blockIdx.x * 256 + threadIdx.x;
    if (i < n) out[i] = f2bf(in[i]);
}

// ---------------- RMSNorm (T5 style) + bf16 cast ----------------
__global__ __launch_bounds__(256)
void rmsnorm_kernel(const float* __restrict__ x, const float* __restrict__ g,
                    u16* __restrict__ out) {
    const int row = blockIdx.x;          // B*L rows
    const int t   = threadIdx.x;         // 256 threads, 4 elems each (D=1024)
    const float* xr = x + (size_t)row * Dc;

    float v[4];
    float ss = 0.0f;
#pragma unroll
    for (int i = 0; i < 4; ++i) { v[i] = xr[t * 4 + i]; ss += v[i] * v[i]; }

#pragma unroll
    for (int m = 1; m < 32; m <<= 1) ss += __shfl_xor(ss, m, 32);

    __shared__ float red[8];
    if ((t & 31) == 0) red[t >> 5] = ss;
    __syncthreads();
    if (t < 8) {
        float z = red[t];
#pragma unroll
        for (int m = 1; m < 8; m <<= 1) z += __shfl_xor(z, m, 32);
        if (t == 0) red[0] = z;
    }
    __syncthreads();

    const float rs = rsqrtf(red[0] / (float)Dc + 1e-6f);
    u16* orow = out + (size_t)row * Dc;
#pragma unroll
    for (int i = 0; i < 4; ++i)
        orow[t * 4 + i] = f2bf(v[i] * rs * g[t * 4 + i]);
}

// ---------------- T5 relative-position bias table ----------------
// biasTab[h][delta + (L-1)] = rel_table[bucket(delta)][h], delta = mem - ctx
__global__ void bias_table_kernel(const float* __restrict__ table, float* __restrict__ biasTab) {
    int idx = blockIdx.x * 256 + threadIdx.x;
    if (idx >= NDELTA * Hc) return;
    int h  = idx & (Hc - 1);
    int di = idx >> 4;                   // 0 .. 4094
    int rp = di - (Lc - 1);              // delta
    int b  = (rp > 0) ? 16 : 0;          // num_buckets/2 = 16
    int rpa = rp < 0 ? -rp : rp;
    if (rpa < 8) {                       // max_exact = 8
        b += rpa;
    } else {
        // 8 + log(rpa/8)/log(16) * 8, clamped to 15
        int lg = 8 + (int)(__logf((float)rpa * 0.125f) * (8.0f / 2.7725887f));
        b += (lg < 15 ? lg : 15);
    }
    biasTab[h * NDELTA + di] = table[b * Hc + h];
}

// ---------------- tiled bf16 WMMA GEMM ----------------
// C[M,N] = A[M,K] * B[K,N]; A,B bf16 row-major.
// MODE 0: store bf16     MODE 1: relu -> bf16     MODE 2: outF = resid + acc
template <int MODE>
__global__ __launch_bounds__(256)
void gemm_bf16_kernel(const u16* __restrict__ A, const u16* __restrict__ B,
                      int M, int N, int Kd,
                      u16* __restrict__ outB,
                      const float* __restrict__ resid, float* __restrict__ outF) {
    __shared__ u16 sA[128][32];      // A tile, row-major
    __shared__ u16 sBt[128][32];     // B tile, stored transposed: [n][k]

    const int t   = threadIdx.x;
    const int l   = t & 31;
    const int w   = t >> 5;          // 8 waves
    const int lm  = l & 15;
    const int hi2 = l >> 4;          // 0/1 half-wave
    const int wm  = w >> 1;          // 0..3  (row group of 32)
    const int wn  = w & 1;           // 0..1  (col group of 64)
    const int mBase = blockIdx.y * 128;
    const int nBase = blockIdx.x * 128;

    v8f acc[2][4];
#pragma unroll
    for (int i = 0; i < 2; ++i)
#pragma unroll
        for (int j = 0; j < 4; ++j) acc[i][j] = (v8f){};

    const int ar = t >> 1, ac = (t & 1) * 16;   // A staging coords
    const int bk = t >> 3, bn = (t & 7) * 16;   // B staging coords
    const int kb8 = hi2 * 8, kb16 = hi2 * 16;

    for (int kt = 0; kt < Kd; kt += 32) {
        // --- stage A (128x32) via CDNA5 async copy engine (no VGPR roundtrip) ---
        const u16* ap = A + (size_t)(mBase + ar) * Kd + kt + ac;
        async_copy_b128(&sA[ar][ac],     ap);
        async_copy_b128(&sA[ar][ac + 8], ap + 8);
        // --- stage B transposed (32x128 -> sBt[n][k]); transpose needs VGPRs ---
        const u16* bp = B + (size_t)(kt + bk) * N + nBase + bn;
        U16B b0 = *reinterpret_cast<const U16B*>(bp);
        U16B b1 = *reinterpret_cast<const U16B*>(bp + 8);
        const u16* s0 = reinterpret_cast<const u16*>(&b0);
        const u16* s1 = reinterpret_cast<const u16*>(&b1);
#pragma unroll
        for (int j = 0; j < 8; ++j) {
            sBt[bn + j][bk]     = s0[j];
            sBt[bn + 8 + j][bk] = s1[j];
        }
        if (kt + 32 < Kd) {   // gfx1250 global_prefetch_b8 of next tiles
            __builtin_prefetch(ap + 32, 0, 1);
            __builtin_prefetch(bp + (size_t)32 * N, 0, 1);
        }
        wait_async0();
        __syncthreads();

        Frag Af[2], Bf[4];
#pragma unroll
        for (int rg = 0; rg < 2; ++rg) {
            const u16* p = &sA[wm * 32 + rg * 16 + lm][0];
            Af[rg].q[0] = *reinterpret_cast<const U16B*>(p + kb8);
            Af[rg].q[1] = *reinterpret_cast<const U16B*>(p + 16 + kb8);
        }
#pragma unroll
        for (int cg = 0; cg < 4; ++cg) {
            const u16* p = &sBt[wn * 64 + cg * 16 + lm][0];
            Bf[cg].q[0] = *reinterpret_cast<const U16B*>(p + kb8);
            Bf[cg].q[1] = *reinterpret_cast<const U16B*>(p + 16 + kb8);
        }
#pragma unroll
        for (int rg = 0; rg < 2; ++rg)
#pragma unroll
            for (int cg = 0; cg < 4; ++cg)
                acc[rg][cg] = __builtin_amdgcn_wmma_f32_16x16x32_bf16(
                    false, Af[rg].v, false, Bf[cg].v, (short)0, acc[rg][cg], false, false);
        __syncthreads();
    }

    // --- epilogue ---
#pragma unroll
    for (int rg = 0; rg < 2; ++rg)
#pragma unroll
        for (int r = 0; r < 8; ++r) {
            const int row = mBase + wm * 32 + rg * 16 + hi2 * 8 + r;
#pragma unroll
            for (int cg = 0; cg < 4; ++cg) {
                const int col = nBase + wn * 64 + cg * 16 + lm;
                const size_t idx = (size_t)row * N + col;
                const float vv = acc[rg][cg][r];
                if (MODE == 0)      outB[idx] = f2bf(vv);
                else if (MODE == 1) outB[idx] = f2bf(fmaxf(vv, 0.0f));
                else                outF[idx] = resid[idx] + vv;
            }
        }
}

// ---------------- flash attention with T5 bias (no 1/sqrt(d)) ----------------
// grid: (L/128, H, B); block 256 = 8 waves; wave w owns 16 q-rows.
__global__ __launch_bounds__(256)
void attn_kernel(const u16* __restrict__ Q, const u16* __restrict__ K,
                 const u16* __restrict__ V, const float* __restrict__ biasTab,
                 u16* __restrict__ Oout) {
    __shared__ u16 sK[32][64];        // K tile, row-major [k'][d]
    __shared__ u16 sVt[64][32];       // V tile transposed [d][k']
    __shared__ u16 sP[8][16][32];     // per-wave P staging (C-layout -> A-layout)

    const int t   = threadIdx.x;
    const int l   = t & 31;
    const int w   = t >> 5;
    const int lm  = l & 15;
    const int hi2 = l >> 4;
    const int h   = blockIdx.y;
    const int bb  = blockIdx.z;
    const int qBase = blockIdx.x * 128 + w * 16;
    const size_t rowStart = (size_t)bb * Lc * Dc;
    const int headOff = h * HDc;
    const int kb8 = hi2 * 8, kb16 = hi2 * 16;

    // Q fragments (A layout), reused across the whole k loop
    Frag Qf[2];
    {
        const u16* qrow = Q + rowStart + (size_t)(qBase + lm) * Dc + headOff;
#pragma unroll
        for (int c = 0; c < 2; ++c) {
            Qf[c].q[0] = *reinterpret_cast<const U16B*>(qrow + c * 32 + kb8);
            Qf[c].q[1] = *reinterpret_cast<const U16B*>(qrow + c * 32 + 16 + kb8);
        }
    }

    v8f O0 = {}, O1 = {}, O2 = {}, O3 = {};
    float mrow[8], lsum[8];
#pragma unroll
    for (int r = 0; r < 8; ++r) { mrow[r] = -3.0e38f; lsum[r] = 0.0f; }

    const float* biasRow = biasTab + h * NDELTA + (Lc - 1);  // index by delta

    const int srow = t >> 3;            // staging: 0..31
    const int sc0  = (t & 7) * 8;       // staging: 0,8,..,56

    for (int kb = 0; kb < Lc / 32; ++kb) {
        // --- stage K (32x64) async, V^T (64x32) via VGPR transpose ---
        {
            const size_t gro = rowStart + (size_t)(kb * 32 + srow) * Dc + headOff + sc0;
            async_copy_b128(&sK[srow][sc0], K + gro);
            U16B vv = *reinterpret_cast<const U16B*>(V + gro);
            const u16* vs = reinterpret_cast<const u16*>(&vv);
#pragma unroll
            for (int j = 0; j < 8; ++j) sVt[sc0 + j][srow] = vs[j];
        }
        wait_async0();
        __syncthreads();

        // --- S = Q K^T : 16x32, two 16-column tiles ---
        v8f S0 = {}, S1 = {};
#pragma unroll
        for (int tt = 0; tt < 2; ++tt) {
            v8f acc = {};
#pragma unroll
            for (int c = 0; c < 2; ++c) {
                Frag Kf;
                const u16* kr = &sK[tt * 16 + lm][c * 32];
                Kf.q[0] = *reinterpret_cast<const U16B*>(kr + kb8);
                Kf.q[1] = *reinterpret_cast<const U16B*>(kr + 16 + kb8);
                acc = __builtin_amdgcn_wmma_f32_16x16x32_bf16(
                    false, Qf[c].v, false, Kf.v, (short)0, acc, false, false);
            }
            if (tt == 0) S0 = acc; else S1 = acc;
        }

        // --- bias + online softmax (per-row, rows live in 16-lane groups) ---
#pragma unroll
        for (int r = 0; r < 8; ++r) {
            const int qrow = qBase + hi2 * 8 + r;
            const int k0   = kb * 32 + lm;
            float s0 = S0[r] + biasRow[k0 - qrow];
            float s1 = S1[r] + biasRow[k0 + 16 - qrow];
            float vmax = fmaxf(s0, s1);
#pragma unroll
            for (int msk = 1; msk < 16; msk <<= 1) vmax = fmaxf(vmax, __shfl_xor(vmax, msk, 32));
            const float mnew  = fmaxf(mrow[r], vmax);
            const float scale = __expf(mrow[r] - mnew);
            mrow[r] = mnew;
            const float p0 = __expf(s0 - mnew);
            const float p1 = __expf(s1 - mnew);
            float ps = p0 + p1;
#pragma unroll
            for (int msk = 1; msk < 16; msk <<= 1) ps += __shfl_xor(ps, msk, 32);
            lsum[r] = lsum[r] * scale + ps;
            O0[r] *= scale; O1[r] *= scale; O2[r] *= scale; O3[r] *= scale;
            const int m = hi2 * 8 + r;
            sP[w][m][lm]      = f2bf(p0);
            sP[w][m][16 + lm] = f2bf(p1);
        }
        // wave-local LDS roundtrip: wait LDS stores, keep scheduler honest
        asm volatile("s_wait_dscnt 0x0" ::: "memory");
        __builtin_amdgcn_wave_barrier();

        // --- P (A layout) from LDS, then O += P V ---
        Frag Pf;
        {
            const u16* pr = &sP[w][lm][0];
            Pf.q[0] = *reinterpret_cast<const U16B*>(pr + kb8);
            Pf.q[1] = *reinterpret_cast<const U16B*>(pr + 16 + kb8);
        }
#pragma unroll
        for (int cg = 0; cg < 4; ++cg) {
            Frag Vf;
            const u16* vr = &sVt[cg * 16 + lm][0];
            Vf.q[0] = *reinterpret_cast<const U16B*>(vr + kb8);
            Vf.q[1] = *reinterpret_cast<const U16B*>(vr + 16 + kb8);
            v8f& Oc = (cg == 0) ? O0 : (cg == 1) ? O1 : (cg == 2) ? O2 : O3;
            Oc = __builtin_amdgcn_wmma_f32_16x16x32_bf16(
                false, Pf.v, false, Vf.v, (short)0, Oc, false, false);
        }
        __syncthreads();
    }

    // --- normalize and store bf16 attention output ---
    u16* obase = Oout + rowStart + headOff;
#pragma unroll
    for (int r = 0; r < 8; ++r) {
        const float inv = 1.0f / lsum[r];
        const size_t qoff = (size_t)(qBase + hi2 * 8 + r) * Dc;
        obase[qoff + 0 * 16 + lm] = f2bf(O0[r] * inv);
        obase[qoff + 1 * 16 + lm] = f2bf(O1[r] * inv);
        obase[qoff + 2 * 16 + lm] = f2bf(O2[r] * inv);
        obase[qoff + 3 * 16 + lm] = f2bf(O3[r] * inv);
    }
}

// ---------------- host-side orchestration ----------------
extern "C" void kernel_launch(void* const* d_in, const int* in_sizes, int n_in,
                              void* d_out, int out_size, void* d_ws, size_t ws_size,
                              hipStream_t stream) {
    (void)in_sizes; (void)n_in; (void)out_size; (void)ws_size;

    const float* x    = (const float*)d_in[0];   // (B,L,D)
    const float* Wq   = (const float*)d_in[1];
    const float* Wk   = (const float*)d_in[2];
    const float* Wv   = (const float*)d_in[3];
    const float* Wo   = (const float*)d_in[4];
    const float* W1   = (const float*)d_in[5];
    const float* W2   = (const float*)d_in[6];
    const float* ln1g = (const float*)d_in[7];
    const float* ln2g = (const float*)d_in[8];
    const float* relt = (const float*)d_in[9];

    constexpr size_t DD   = (size_t)Dc * Dc;     // 1M
    constexpr size_t DF   = (size_t)Dc * DFFc;   // 4M
    constexpr size_t BLD  = (size_t)Bc * Lc * Dc;   // 4M
    constexpr size_t BLF  = (size_t)Bc * Lc * DFFc; // 16M

    u16* WqB   = (u16*)d_ws;
    u16* WkB   = WqB + DD;
    u16* WvB   = WkB + DD;
    u16* WoB   = WvB + DD;
    u16* W1B   = WoB + DD;
    u16* W2B   = W1B + DF;
    u16* h1B   = W2B + DF;
    u16* QB    = h1B + BLD;
    u16* KB    = QB  + BLD;
    u16* VB    = KB  + BLD;
    u16* attnB = VB  + BLD;
    u16* h2B   = attnB + BLD;
    u16* midB  = h2B + BLD;
    float* x1F   = (float*)(midB + BLF);
    float* biasF = x1F + BLD;

    const int ROWS = Bc * Lc;   // 4096

    // 1) weights -> bf16
    cast_bf16_kernel<<<(int)((DD + 255) / 256), 256, 0, stream>>>(Wq, WqB, (int)DD);
    cast_bf16_kernel<<<(int)((DD + 255) / 256), 256, 0, stream>>>(Wk, WkB, (int)DD);
    cast_bf16_kernel<<<(int)((DD + 255) / 256), 256, 0, stream>>>(Wv, WvB, (int)DD);
    cast_bf16_kernel<<<(int)((DD + 255) / 256), 256, 0, stream>>>(Wo, WoB, (int)DD);
    cast_bf16_kernel<<<(int)((DF + 255) / 256), 256, 0, stream>>>(W1, W1B, (int)DF);
    cast_bf16_kernel<<<(int)((DF + 255) / 256), 256, 0, stream>>>(W2, W2B, (int)DF);

    // 2) pre-norm 1 (bf16) + bias table
    rmsnorm_kernel<<<ROWS, 256, 0, stream>>>(x, ln1g, h1B);
    bias_table_kernel<<<(NDELTA * Hc + 255) / 256, 256, 0, stream>>>(relt, biasF);

    // 3) Q/K/V projections (M=4096, N=1024, K=1024)
    dim3 gQKV(Dc / 128, ROWS / 128);
    gemm_bf16_kernel<0><<<gQKV, 256, 0, stream>>>(h1B, WqB, ROWS, Dc, Dc, QB, nullptr, nullptr);
    gemm_bf16_kernel<0><<<gQKV, 256, 0, stream>>>(h1B, WkB, ROWS, Dc, Dc, KB, nullptr, nullptr);
    gemm_bf16_kernel<0><<<gQKV, 256, 0, stream>>>(h1B, WvB, ROWS, Dc, Dc, VB, nullptr, nullptr);

    // 4) flash attention with relative-position bias
    attn_kernel<<<dim3(Lc / 128, Hc, Bc), 256, 0, stream>>>(QB, KB, VB, biasF, attnB);

    // 5) output projection + residual -> x1 (f32)
    gemm_bf16_kernel<2><<<gQKV, 256, 0, stream>>>(attnB, WoB, ROWS, Dc, Dc, nullptr, x, x1F);

    // 6) pre-norm 2 + FFN
    rmsnorm_kernel<<<ROWS, 256, 0, stream>>>(x1F, ln2g, h2B);
    dim3 gFF1(DFFc / 128, ROWS / 128);
    gemm_bf16_kernel<1><<<gFF1, 256, 0, stream>>>(h2B, W1B, ROWS, DFFc, Dc, midB, nullptr, nullptr);
    gemm_bf16_kernel<2><<<gQKV, 256, 0, stream>>>(midB, W2B, ROWS, Dc, DFFc, nullptr, x1F, (float*)d_out);
}
